// DeTree_51977694216747
// MI455X (gfx1250) — compile-verified, matrix operand order checked by reading, not run on hardware
//
#include <hip/hip_runtime.h>
#include <hip/hip_bf16.h>

// Problem constants (match reference)
#define F_IN    512            // in_features (GEMM K)
#define N_TREES 512
#define TDEPTH  5
#define NLEAF   32
#define BATCH   2048           // GEMM M
#define ND      (N_TREES * TDEPTH)   // 2560, GEMM N

typedef _Float16 v16h __attribute__((ext_vector_type(16)));
typedef _Float16 v8h  __attribute__((ext_vector_type(8)));
typedef float    v8f  __attribute__((ext_vector_type(8)));

#define WMMA_F16(A, B, C) \
    __builtin_amdgcn_wmma_f32_16x16x32_f16(false, (A), false, (B), (short)0, (C), false, false)

// ---------------------------------------------------------------------------
// Kernel 1: x (f32) -> f16, row-major [BATCH][F_IN]  (WMMA A operand)
// ---------------------------------------------------------------------------
__global__ __launch_bounds__(256) void cvt_x_kernel(const float* __restrict__ x,
                                                    _Float16* __restrict__ xh) {
    int i = blockIdx.x * 256 + threadIdx.x;     // exact grid: BATCH*F_IN/256
    xh[i] = (_Float16)x[i];
}

// ---------------------------------------------------------------------------
// Kernel 2: column softmax of feat_attention [F_IN rows][ND cols] over rows,
// written TRANSPOSED as f16:  wt[c * F_IN + f]   (WMMA B operand, K-contig)
// One wave32 per column, 16 values per lane, shuffle reductions.
// ---------------------------------------------------------------------------
__global__ __launch_bounds__(256) void softmax_cols_kernel(const float* __restrict__ fa,
                                                           _Float16* __restrict__ wt) {
    int col  = (blockIdx.x * 256 + threadIdx.x) >> 5;   // 0..ND-1
    int lane = threadIdx.x & 31;
    if (col >= ND) return;

    float v[16];
    float m = -3.4e38f;
#pragma unroll
    for (int i = 0; i < 16; ++i) {
        v[i] = fa[(size_t)(lane + 32 * i) * ND + col];
        m = fmaxf(m, v[i]);
    }
#pragma unroll
    for (int off = 16; off > 0; off >>= 1) m = fmaxf(m, __shfl_xor(m, off, 32));

    float s = 0.0f;
#pragma unroll
    for (int i = 0; i < 16; ++i) { v[i] = __expf(v[i] - m); s += v[i]; }
#pragma unroll
    for (int off = 16; off > 0; off >>= 1) s += __shfl_xor(s, off, 32);

    float inv = 1.0f / s;
#pragma unroll
    for (int i = 0; i < 16; ++i)
        wt[(size_t)col * F_IN + lane + 32 * i] = (_Float16)(v[i] * inv);
}

// ---------------------------------------------------------------------------
// Kernel 3: WMMA GEMM  FV[BATCH][ND] = Xh[BATCH][F_IN] * Wt^T
// CDNA5 16-bit A/B fragment layout (ISA 7.12.2):
//   lane<16  : row/col = lane,    K = {0..7} U {16..23}
//   lane>=16 : row/col = lane-16, K = {8..15} U {24..31}
// -> two aligned 16-byte loads per fragment.
// Each wave: 2 M-tiles x 4 N-tiles (8 WMMA per k-step), with register
// double-buffering of next k-step fragments so loads overlap WMMAs.
// ---------------------------------------------------------------------------
__device__ __forceinline__ v16h load_frag16(const _Float16* __restrict__ base,
                                            int k0, int hi) {
    const _Float16* p = base + k0 + hi * 8;
    v8h lo = *(const v8h*)(p);        // K = k0+hi*8 .. +7
    v8h up = *(const v8h*)(p + 16);   // K = k0+16+hi*8 .. +7
    v16h r;
#pragma unroll
    for (int i = 0; i < 8; ++i) { r[i] = lo[i]; r[i + 8] = up[i]; }
    return r;
}

__global__ __launch_bounds__(256) void gemm_fv_kernel(const _Float16* __restrict__ xh,
                                                      const _Float16* __restrict__ wt,
                                                      float* __restrict__ fv) {
    const int NGROUP = ND / 64;                       // 40 groups of 4 N-tiles
    int wid  = blockIdx.x * 8 + (threadIdx.x >> 5);   // global wave id
    int lane = threadIdx.x & 31;
    int mt = wid / NGROUP;                            // 0..63 (32-row supertile)
    int ng = wid % NGROUP;                            // 0..39
    int m_base = mt * 32;
    int n_base = ng * 64;
    int hl = lane & 15;
    int hi = lane >> 4;

    const _Float16* arow0 = xh + (size_t)(m_base + hl) * F_IN;
    const _Float16* arow1 = arow0 + (size_t)16 * F_IN;
    const _Float16* bcol  = wt + (size_t)(n_base + hl) * F_IN;

    v8f acc00 = {}, acc01 = {}, acc02 = {}, acc03 = {};
    v8f acc10 = {}, acc11 = {}, acc12 = {}, acc13 = {};

    // Prologue: fragments for k-step 0
    v16h a0 = load_frag16(arow0, 0, hi);
    v16h a1 = load_frag16(arow1, 0, hi);
    v16h b0 = load_frag16(bcol,              0, hi);
    v16h b1 = load_frag16(bcol + 16 * F_IN,  0, hi);
    v16h b2 = load_frag16(bcol + 32 * F_IN,  0, hi);
    v16h b3 = load_frag16(bcol + 48 * F_IN,  0, hi);

#pragma unroll
    for (int kk = 0; kk < F_IN; kk += 32) {
        // Issue next k-step loads BEFORE consuming current fragments.
        v16h na0 = a0, na1 = a1, nb0 = b0, nb1 = b1, nb2 = b2, nb3 = b3;
        if (kk + 32 < F_IN) {
            na0 = load_frag16(arow0, kk + 32, hi);
            na1 = load_frag16(arow1, kk + 32, hi);
            nb0 = load_frag16(bcol,              kk + 32, hi);
            nb1 = load_frag16(bcol + 16 * F_IN,  kk + 32, hi);
            nb2 = load_frag16(bcol + 32 * F_IN,  kk + 32, hi);
            nb3 = load_frag16(bcol + 48 * F_IN,  kk + 32, hi);
        }
        acc00 = WMMA_F16(a0, b0, acc00);
        acc01 = WMMA_F16(a0, b1, acc01);
        acc02 = WMMA_F16(a0, b2, acc02);
        acc03 = WMMA_F16(a0, b3, acc03);
        acc10 = WMMA_F16(a1, b0, acc10);
        acc11 = WMMA_F16(a1, b1, acc11);
        acc12 = WMMA_F16(a1, b2, acc12);
        acc13 = WMMA_F16(a1, b3, acc13);
        a0 = na0; a1 = na1; b0 = nb0; b1 = nb1; b2 = nb2; b3 = nb3;
    }

    // C/D layout: VGPR r -> M = r + 8*hi, N = lane&15
    int row0 = m_base + hi * 8;
#pragma unroll
    for (int r = 0; r < 8; ++r) {
        float* o0 = fv + (size_t)(row0 + r) * ND + n_base + hl;
        o0[0]  = acc00[r];
        o0[16] = acc01[r];
        o0[32] = acc02[r];
        o0[48] = acc03[r];
        float* o1 = o0 + (size_t)16 * ND;
        o1[0]  = acc10[r];
        o1[16] = acc11[r];
        o1[32] = acc12[r];
        o1[48] = acc13[r];
    }
}

// ---------------------------------------------------------------------------
// Kernel 4: sparsemoid gates + oblivious-tree leaf fold + response dot.
// path_map semantics: at depth j, gate bit = (leaf >> j) & 1  (closed form).
// c0 = 1 - c1 holds identically under the clamp.  Binary fold: 62 mults.
// ---------------------------------------------------------------------------
__global__ __launch_bounds__(256) void tree_epilogue_kernel(const float* __restrict__ fv,
                                                            const float* __restrict__ thr,
                                                            const float* __restrict__ logt,
                                                            const float* __restrict__ resp,
                                                            float* __restrict__ out) {
    int idx = blockIdx.x * 256 + threadIdx.x;   // exact grid: BATCH*N_TREES/256
    int b = idx >> 9;          // / N_TREES (512)
    int n = idx & (N_TREES - 1);

    float c1[TDEPTH];
#pragma unroll
    for (int j = 0; j < TDEPTH; ++j) {
        float x = fv[(size_t)b * ND + n * TDEPTH + j];
        float t = (x - thr[n * TDEPTH + j]) * __expf(-logt[n * TDEPTH + j]);
        c1[j] = fminf(fmaxf(0.5f * t + 0.5f, 0.0f), 1.0f);
    }

    float buf[NLEAF];
    const float* r = resp + (size_t)n * NLEAF;   // response[n, 0, :]
#pragma unroll
    for (int l = 0; l < NLEAF; ++l) buf[l] = r[l];

#pragma unroll
    for (int j = 0; j < TDEPTH; ++j) {
        float c = c1[j];
        float c0 = 1.0f - c;
        int len = NLEAF >> (j + 1);
#pragma unroll
        for (int m = 0; m < 16; ++m) {
            if (m < len) buf[m] = buf[2 * m] * c0 + buf[2 * m + 1] * c;
        }
    }
    out[idx] = buf[0];
}

// ---------------------------------------------------------------------------
extern "C" void kernel_launch(void* const* d_in, const int* in_sizes, int n_in,
                              void* d_out, int out_size, void* d_ws, size_t ws_size,
                              hipStream_t stream) {
    (void)in_sizes; (void)n_in; (void)out_size; (void)ws_size;

    const float* x    = (const float*)d_in[0];   // [2048, 512]
    const float* fa   = (const float*)d_in[1];   // [512, 2560]
    const float* thr  = (const float*)d_in[2];   // [512, 5]
    const float* logt = (const float*)d_in[3];   // [512, 5]
    const float* resp = (const float*)d_in[4];   // [512, 1, 32]
    // d_in[5] = path_map (int) — encoded in closed form in the epilogue.

    char* ws = (char*)d_ws;
    size_t off = 0;
    _Float16* xh = (_Float16*)(ws + off); off += (size_t)BATCH * F_IN * sizeof(_Float16); // 2 MB
    _Float16* wt = (_Float16*)(ws + off); off += (size_t)ND * F_IN * sizeof(_Float16);    // 2.5 MB
    float*    fv = (float*)(ws + off);                                                    // 20 MB

    float* out = (float*)d_out;                  // [2048, 512]

    cvt_x_kernel<<<(BATCH * F_IN) / 256, 256, 0, stream>>>(x, xh);
    softmax_cols_kernel<<<ND / 8, 256, 0, stream>>>(fa, wt);                   // 320 blocks
    gemm_fv_kernel<<<(BATCH / 32) * (ND / 64) / 8, 256, 0, stream>>>(xh, wt, fv); // 320 blocks
    tree_epilogue_kernel<<<(BATCH * N_TREES) / 256, 256, 0, stream>>>(fv, thr, logt, resp, out);
}
